// IdealDenoiser_25718264169333
// MI455X (gfx1250) — matrix operand order repaired and checked by reading, not compiled, not run on hardware
//
#include <hip/hip_runtime.h>
#include <hip/hip_bf16.h>

// ---------------------------------------------------------------------------
// IdealDenoiser on MI455X (gfx1250).
//   x:     [256, 3072] f32      (d_in[0])
//   sigma: scalar f32           (d_in[1])
//   data:  [50000, 3072] f32    (d_in[2])
//   out:   [256, 3072] f32
//
// Two bf16-WMMA GEMMs (fp32 accumulate) around a softmax over N:
//   logitsT[b,n] = (2 d.x - |d|^2 - |x|^2) / (2 sigma^2)   (51 MB, L2)
//   W[b,n]       = softmax_n(logitsT)  (bf16, zero-padded to NPAD)
//   eps          = W @ data   (K = n, 4 deterministic K-slices + combine)
//   out          = (x - eps) / sigma
// ---------------------------------------------------------------------------

#define NB     256        // batch (columns)
#define NN     50000      // dataset rows
#define DD     3072       // feature dim
#define NPAD   50176      // NN rounded up to 4 slices * 98 stages * 128 rows
#define KC     1024       // k_logits LDS stage width (3 stages over DD)
#define NSLICE 4          // k_eps K-slices
#define NCHUNK 128        // k_eps rows staged per barrier (4 WMMA steps)
#define SLICEN (NPAD / NSLICE)   // 12544 rows per slice

typedef __attribute__((ext_vector_type(16))) __bf16 v16bf;
typedef __attribute__((ext_vector_type(8)))  float  v8f;

union FragAB { v16bf v; unsigned u[8]; };

__device__ __forceinline__ unsigned short f2bf(float f) {
    union { float f; unsigned u; } x; x.f = f;
    unsigned r = x.u + 0x7FFFu + ((x.u >> 16) & 1u);   // round-to-nearest-even
    return (unsigned short)(r >> 16);
}

// ---------------------------------------------------------------------------
// Kernel 0: x -> bf16 (row major) and xnorm[b] = sum_d x^2
// ---------------------------------------------------------------------------
__global__ __launch_bounds__(256)
void k_prep_x(const float* __restrict__ x,
              unsigned short* __restrict__ x_bf,
              float* __restrict__ xnorm)
{
    __shared__ float red[256];
    const int b = blockIdx.x;
    const int t = threadIdx.x;
    float acc = 0.f;
    for (int d = t; d < DD; d += 256) {
        float v = x[b * DD + d];
        x_bf[b * DD + d] = f2bf(v);
        acc += v * v;
    }
    red[t] = acc;
    __syncthreads();
    for (int o = 128; o > 0; o >>= 1) {
        if (t < o) red[t] += red[t + o];
        __syncthreads();
    }
    if (t == 0) xnorm[b] = red[0];
}

// ---------------------------------------------------------------------------
// Kernel 1: logitsT[b, n] = (2 d.x - |d|^2 - |x|^2) / (2 sigma^2)
// grid = 3125 (16-row data tiles), block = 512 = 16 waves.
// Wave w owns b-tile [16w, 16w+16). The 16x1024 bf16 data chunk (A operand)
// is staged cooperatively in LDS (float4 loads -> packed ds_store_b64); then
// 32 WMMA K-steps run per barrier pair. |d|^2 folds into the staging pass.
// ---------------------------------------------------------------------------
__global__ __launch_bounds__(512)
void k_logits(const float* __restrict__ data,
              const unsigned short* __restrict__ x_bf,
              const float* __restrict__ xnorm,
              const float* __restrict__ sigma,
              float* __restrict__ logitsT)
{
    __shared__ unsigned short Atile[16 * KC];   // [n(16)][k(KC)] bf16 (32 KB)
    __shared__ float dnorm_s[16];

    const int tid  = threadIdx.x;
    const int wave = tid >> 5;          // 0..15 (also the data row this thread stages)
    const int lane = tid & 31;
    const int n0   = blockIdx.x * 16;

    const int m    = lane & 15;
    const int kbA  = (lane < 16) ? 0 : 16;   // A-operand K base
    const int kbB  = (lane < 16) ? 0 : 16;   // B-operand K base
    const int bcol = (wave << 4) + m;        // this lane's batch column

    const unsigned* A32 = (const unsigned*)Atile;
    const unsigned* X32 = (const unsigned*)x_bf;
    const float4*   drow4 = (const float4*)(data + (size_t)(n0 + wave) * DD);
    unsigned long long* A64 = (unsigned long long*)Atile;

    float dn = 0.f;                           // partial |d_row|^2 (row = wave)
    v8f acc = {0.f, 0.f, 0.f, 0.f, 0.f, 0.f, 0.f, 0.f};

    for (int c0 = 0; c0 < DD; c0 += KC) {
        __syncthreads();                      // previous chunk fully consumed
        #pragma unroll
        for (int i = 0; i < KC / 128; ++i) {  // 8 x (b128 load + cvt + ds_store_b64)
            float4 v = drow4[(c0 >> 2) + lane + 32 * i];
            dn += v.x * v.x + v.y * v.y + v.z * v.z + v.w * v.w;
            unsigned lo = (unsigned)f2bf(v.x) | ((unsigned)f2bf(v.y) << 16);
            unsigned hi = (unsigned)f2bf(v.z) | ((unsigned)f2bf(v.w) << 16);
            A64[(wave * KC + lane * 4 + 128 * i) >> 2] =
                ((unsigned long long)hi << 32) | lo;
        }
        __syncthreads();                      // chunk ready

        #pragma unroll 4
        for (int ks = 0; ks < KC; ks += 32) {
            FragAB a, bfrag;
            #pragma unroll
            for (int p = 0; p < 8; ++p) {
                int K = kbA + 2 * p;
                a.u[p] = A32[(m * KC + ks + K) >> 1];
            }
            #pragma unroll
            for (int p = 0; p < 8; ++p) {
                int K = kbB + 2 * p;
                bfrag.u[p] = X32[((size_t)bcol * DD + c0 + ks + K) >> 1];
            }
            acc = __builtin_amdgcn_wmma_f32_16x16x32_bf16(
                      false, a.v, false, bfrag.v, (short)0, acc, false, false);
        }
    }

    // reduce |d|^2 across the wave (all 32 lanes hold partials of row `wave`)
    #pragma unroll
    for (int o = 16; o > 0; o >>= 1) dn += __shfl_xor(dn, o, 32);
    if (lane == 0) dnorm_s[wave] = dn;
    __syncthreads();

    const float sg   = sigma[0];
    const float i2s2 = 0.5f / (sg * sg);
    const int   b0   = wave << 4;

    #pragma unroll
    for (int v = 0; v < 8; ++v) {
        int M  = v + ((lane < 16) ? 0 : 8);
        int Nc = lane & 15;
        float l = (2.f * acc[v] - dnorm_s[M] - xnorm[b0 + Nc]) * i2s2;
        logitsT[(size_t)(b0 + Nc) * NPAD + (n0 + M)] = l;
    }
}

// ---------------------------------------------------------------------------
// Kernel 2: per-column softmax stats over N (max, sum-exp). Fully coalesced
// on logitsT. grid = 256 (one block per column), block = 256.
// ---------------------------------------------------------------------------
__global__ __launch_bounds__(256)
void k_stats(const float* __restrict__ logitsT,
             float* __restrict__ colmax,
             float* __restrict__ colsum)
{
    __shared__ float red[256];
    const int b = blockIdx.x;
    const int t = threadIdx.x;
    const float* row = logitsT + (size_t)b * NPAD;

    float m = -3.4e38f;
    for (int n = t; n < NN; n += 256) m = fmaxf(m, row[n]);
    red[t] = m;
    __syncthreads();
    for (int o = 128; o > 0; o >>= 1) {
        if (t < o) red[t] = fmaxf(red[t], red[t + o]);
        __syncthreads();
    }
    m = red[0];
    __syncthreads();

    float s = 0.f;
    for (int n = t; n < NN; n += 256) s += __expf(row[n] - m);
    red[t] = s;
    __syncthreads();
    for (int o = 128; o > 0; o >>= 1) {
        if (t < o) red[t] += red[t + o];
        __syncthreads();
    }
    if (t == 0) { colmax[b] = m; colsum[b] = red[0]; }
}

// ---------------------------------------------------------------------------
// Kernel 3: W[b][n] = exp(l - m)/Z as bf16, zero pad to NPAD. Coalesced.
// grid = (NPAD/256, 256), block = 256.
// ---------------------------------------------------------------------------
__global__ __launch_bounds__(256)
void k_weights(const float* __restrict__ logitsT,
               const float* __restrict__ colmax,
               const float* __restrict__ colsum,
               unsigned short* __restrict__ w_t)
{
    const int n = blockIdx.x * 256 + threadIdx.x;
    const int b = blockIdx.y;
    unsigned short w = 0;
    if (n < NN) {
        float l = logitsT[(size_t)b * NPAD + n];
        w = f2bf(__expf(l - colmax[b]) * (1.f / colsum[b]));
    }
    w_t[(size_t)b * NPAD + n] = w;
}

// ---------------------------------------------------------------------------
// Kernel 4: epsPart[slice] = W[:, slice] @ data[slice, :]
// grid = (192 d-tiles, 4 K-slices), block = 512 = 16 waves; wave w owns
// b-tile w. 128 data rows are staged fp32->bf16 transposed in LDS per
// barrier (4 WMMA steps), with a prefetch of the next stage. Weights (A)
// are dword-gathered straight from L2-resident w_t. Deterministic: each
// slice writes its own partial buffer; no float atomics.
// ---------------------------------------------------------------------------
__global__ __launch_bounds__(512)
void k_eps(const float* __restrict__ data,
           const unsigned short* __restrict__ w_t,
           float* __restrict__ epsPart)
{
    __shared__ unsigned short T[16 * NCHUNK];   // [d(16)][n(128)] bf16 (4 KB)

    const int tid   = threadIdx.x;
    const int wave  = tid >> 5;
    const int lane  = tid & 31;
    const int d0    = blockIdx.x * 16;
    const int slice = blockIdx.y;
    const int b0    = wave << 4;
    const int nbeg  = slice * SLICEN;

    const int rstage = tid >> 4;              // staged n-row   0..31 (x4)
    const int cstage = tid & 15;              // staged d-col   0..15

    const int m    = lane & 15;
    const int kbA  = (lane < 16) ? 0 : 16;
    const int kbB  = (lane < 16) ? 0 : 16;
    const int brow = b0 + m;                  // this lane's A-matrix row (batch col)

    const unsigned* W32 = (const unsigned*)w_t;
    const unsigned* T32 = (const unsigned*)T;

    v8f acc = {0.f, 0.f, 0.f, 0.f, 0.f, 0.f, 0.f, 0.f};

    for (int n0 = nbeg; n0 < nbeg + SLICEN; n0 += NCHUNK) {
        __syncthreads();
        #pragma unroll
        for (int i = 0; i < NCHUNK / 32; ++i) {
            int n = n0 + rstage + 32 * i;
            float v = (n < NN) ? data[(size_t)n * DD + d0 + cstage] : 0.f;
            T[cstage * NCHUNK + rstage + 32 * i] = f2bf(v);
        }
        {   // prefetch the first rows of the next stage (global_prefetch_b8)
            int np = n0 + NCHUNK + rstage;
            if (np < NN)
                __builtin_prefetch(data + (size_t)np * DD + d0 + cstage, 0, 1);
        }
        __syncthreads();

        #pragma unroll
        for (int s = 0; s < NCHUNK / 32; ++s) {
            const int ks = 32 * s;
            FragAB a, bfrag;
            #pragma unroll
            for (int p = 0; p < 8; ++p) {
                int K = kbA + 2 * p;
                a.u[p] = W32[((size_t)brow * NPAD + n0 + ks + K) >> 1];
            }
            #pragma unroll
            for (int p = 0; p < 8; ++p) {
                int K = kbB + 2 * p;
                bfrag.u[p] = T32[(m * NCHUNK + ks + K) >> 1];
            }
            acc = __builtin_amdgcn_wmma_f32_16x16x32_bf16(
                      false, a.v, false, bfrag.v, (short)0, acc, false, false);
        }
    }

    float* part = epsPart + (size_t)slice * NB * DD;
    #pragma unroll
    for (int v = 0; v < 8; ++v) {
        int M  = v + ((lane < 16) ? 0 : 8);
        int Nc = lane & 15;
        part[(size_t)(b0 + M) * DD + d0 + Nc] = acc[v];
    }
}

// ---------------------------------------------------------------------------
// Kernel 5: out = (x - sum_slices epsPart) / sigma
// ---------------------------------------------------------------------------
__global__ __launch_bounds__(256)
void k_combine(const float* __restrict__ x,
               const float* __restrict__ epsPart,
               const float* __restrict__ sigma,
               float* __restrict__ out)
{
    const size_t i = (size_t)blockIdx.x * 256 + threadIdx.x;
    float e = 0.f;
    #pragma unroll
    for (int s = 0; s < NSLICE; ++s)
        e += epsPart[(size_t)s * NB * DD + i];
    out[i] = (x[i] - e) * (1.f / sigma[0]);
}

// ---------------------------------------------------------------------------
// Launch
// ---------------------------------------------------------------------------
extern "C" void kernel_launch(void* const* d_in, const int* in_sizes, int n_in,
                              void* d_out, int out_size, void* d_ws, size_t ws_size,
                              hipStream_t stream)
{
    const float* x     = (const float*)d_in[0];
    const float* sigma = (const float*)d_in[1];
    const float* data  = (const float*)d_in[2];
    float*       out   = (float*)d_out;

    // Workspace layout (bytes):
    //   logitsT f32 [NB, NPAD]   : 51,380,224
    //   w_t     bf16 [NB, NPAD]  : 25,690,112
    //   x_bf    bf16 [NB, DD]    :  1,572,864
    //   epsPart f32 [4, NB, DD]  : 12,582,912
    //   xnorm / colmax / colsum  :  3 * 1,024
    char* ws = (char*)d_ws;
    float*          logitsT = (float*)(ws);
    unsigned short* w_t     = (unsigned short*)(ws + (size_t)NB * NPAD * 4);
    unsigned short* x_bf    = (unsigned short*)(ws + (size_t)NB * NPAD * 6);
    float*          epsPart = (float*)((char*)x_bf + (size_t)NB * DD * 2);
    float*          xnorm   = epsPart + (size_t)NSLICE * NB * DD;
    float*          colmax  = xnorm + NB;
    float*          colsum  = colmax + NB;

    k_prep_x<<<NB, 256, 0, stream>>>(x, x_bf, xnorm);
    k_logits<<<NN / 16, 512, 0, stream>>>(data, x_bf, xnorm, sigma, logitsT);
    k_stats<<<NB, 256, 0, stream>>>(logitsT, colmax, colsum);
    dim3 wgrid(NPAD / 256, NB);
    k_weights<<<wgrid, 256, 0, stream>>>(logitsT, colmax, colsum, w_t);
    dim3 egrid(DD / 16, NSLICE);
    k_eps<<<egrid, 512, 0, stream>>>(data, w_t, epsPart);
    k_combine<<<(NB * DD) / 256, 256, 0, stream>>>(x, epsPart, sigma, out);
}
